// Block_7009386627324
// MI455X (gfx1250) — matrix operand order repaired and checked by reading, not compiled, not run on hardware
//
#include <hip/hip_runtime.h>
#include <hip/hip_bf16.h>
#include <cstdint>
#include <math.h>

// ---------------------------------------------------------------------------
// CDNA5 / gfx1250 transformer block: all GEMMs via v_wmma_f32_16x16x32_bf16
// ---------------------------------------------------------------------------

typedef __attribute__((ext_vector_type(16))) __bf16 v16bf;
typedef __attribute__((ext_vector_type(8)))  float  v8f;

union Frag16 { uint4 u[2]; v16bf v; };

__device__ __forceinline__ uint16_t f2bf(float f) {
  uint32_t u = __float_as_uint(f);
  u += 0x7FFFu + ((u >> 16) & 1u);   // round-to-nearest-even
  return (uint16_t)(u >> 16);
}

// A-operand fragment (16x32 bf16), ISA 7.12.2 layout:
//   lane&15 = M row; lane>>4 selects K-half within each 8-K group.
//   V0-3: K 0-7 (half0) / 8-15 (half1); V4-7: K 16-23 / 24-31.
__device__ __forceinline__ Frag16 ldsFragA(const uint32_t* s, int row0, int stride,
                                           int lane, int cdw) {
  const int half = lane >> 4, m = lane & 15;
  const uint32_t* p = s + (size_t)(row0 + m) * stride + cdw + half * 4;
  Frag16 f;
  f.u[0] = *(const uint4*)(p);
  f.u[1] = *(const uint4*)(p + 8);
  return f;
}

// B-operand fragment (32x16 bf16): lane&15 = N col (row of Bt); same K layout as A:
// V0-3: K 0-7 (half0) / 8-15 (half1); V4-7: K 16-23 / 24-31.
__device__ __forceinline__ Frag16 ldsFragB(const uint32_t* s, int row0, int stride,
                                           int lane, int cdw) {
  const int half = lane >> 4, n = lane & 15;
  const uint32_t* p = s + (size_t)(row0 + n) * stride + cdw + half * 4;
  Frag16 f;
  f.u[0] = *(const uint4*)(p);
  f.u[1] = *(const uint4*)(p + 8);
  return f;
}

// ---------------------------------------------------------------------------
// Weight transpose + f32->bf16 convert:  w[K][N] (f32) -> wt[N][K] (bf16)
// ---------------------------------------------------------------------------
__global__ __launch_bounds__(256)
void transpose_bf16_kernel(const float* __restrict__ w, uint16_t* __restrict__ wt,
                           int K, int N) {
  __shared__ float tile[32][33];
  const int tx = threadIdx.x, ty = threadIdx.y;      // (32, 8)
  const int n0 = blockIdx.x * 32, k0 = blockIdx.y * 32;
  #pragma unroll
  for (int i = 0; i < 4; ++i)
    tile[ty + i * 8][tx] = w[(size_t)(k0 + ty + i * 8) * N + n0 + tx];
  __syncthreads();
  #pragma unroll
  for (int i = 0; i < 4; ++i)
    wt[(size_t)(n0 + ty + i * 8) * K + k0 + tx] = f2bf(tile[tx][ty + i * 8]);
}

// ---------------------------------------------------------------------------
// Row LayerNorm (C=1024) f32 in -> bf16 out
// ---------------------------------------------------------------------------
__global__ __launch_bounds__(256)
void ln_bf16_kernel(const float* __restrict__ x, const float* __restrict__ g,
                    const float* __restrict__ bta, uint16_t* __restrict__ out) {
  const int C = 1024;
  const int row = blockIdx.x, tid = threadIdx.x;
  const float4 v = ((const float4*)(x + (size_t)row * C))[tid];
  float s  = v.x + v.y + v.z + v.w;
  float ss = v.x * v.x + v.y * v.y + v.z * v.z + v.w * v.w;
  #pragma unroll
  for (int o = 1; o < 32; o <<= 1) {
    s  += __shfl_xor(s, o, 32);
    ss += __shfl_xor(ss, o, 32);
  }
  __shared__ float red[16];
  const int wave = tid >> 5, lane = tid & 31;
  if (lane == 0) { red[wave] = s; red[wave + 8] = ss; }
  __syncthreads();
  if (tid == 0) {
    float a = 0.f, b2 = 0.f;
    #pragma unroll
    for (int i = 0; i < 8; ++i) { a += red[i]; b2 += red[i + 8]; }
    red[0] = a; red[8] = b2;
  }
  __syncthreads();
  const float mu   = red[0] * (1.0f / C);
  const float var  = red[8] * (1.0f / C) - mu * mu;
  const float rstd = rsqrtf(var + 1e-5f);
  const float4 gg = ((const float4*)g)[tid];
  const float4 bb = ((const float4*)bta)[tid];
  uint32_t lo = (uint32_t)f2bf((v.x - mu) * rstd * gg.x + bb.x) |
                ((uint32_t)f2bf((v.y - mu) * rstd * gg.y + bb.y) << 16);
  uint32_t hi = (uint32_t)f2bf((v.z - mu) * rstd * gg.z + bb.z) |
                ((uint32_t)f2bf((v.w - mu) * rstd * gg.w + bb.w) << 16);
  uint2 pk; pk.x = lo; pk.y = hi;
  ((uint2*)(out + (size_t)row * C))[tid] = pk;
}

// ---------------------------------------------------------------------------
// bf16 WMMA GEMM: C[M][N] = A[M][K] * Bt[N][K]^T + bias, with epilogues
// ---------------------------------------------------------------------------
#define EPI_QKV       0
#define EPI_RES_F32   1
#define EPI_GELU_BF16 2

template <int EPI>
__global__ __launch_bounds__(256)
void gemm_bf16_kernel(const uint16_t* __restrict__ A, const uint16_t* __restrict__ Bt,
                      const float* __restrict__ bias, int M, int N, int K,
                      float* __restrict__ outF, const float* __restrict__ res,
                      uint16_t* __restrict__ oQ, uint16_t* __restrict__ oK,
                      uint16_t* __restrict__ oV) {
  __shared__ uint32_t sA[128 * 20];  // 128 rows x 32 bf16, stride 20 dwords (pad)
  __shared__ uint32_t sB[128 * 20];
  const int tid = threadIdx.x;
  const int lane = tid & 31, wave = tid >> 5;
  const int wm = wave >> 2, wn = wave & 3;              // 2 x 4 wave grid
  const int m0 = blockIdx.y * 128, n0 = blockIdx.x * 128;

  v8f acc[4][2] = {};
  for (int kt = 0; kt < K; kt += 32) {
    #pragma unroll
    for (int p = 0; p < 2; ++p) {
      const int it = tid + p * 256;
      const int r = it >> 2, c4 = it & 3;
      *(uint4*)&sA[r * 20 + c4 * 4] =
          *(const uint4*)(A + (size_t)(m0 + r) * K + kt + c4 * 8);
      *(uint4*)&sB[r * 20 + c4 * 4] =
          *(const uint4*)(Bt + (size_t)(n0 + r) * K + kt + c4 * 8);
    }
    __syncthreads();
    Frag16 af[4], bfr[2];
    #pragma unroll
    for (int i = 0; i < 4; ++i) af[i] = ldsFragA(sA, wm * 64 + i * 16, 20, lane, 0);
    #pragma unroll
    for (int j = 0; j < 2; ++j) bfr[j] = ldsFragB(sB, wn * 32 + j * 16, 20, lane, 0);
    #pragma unroll
    for (int i = 0; i < 4; ++i)
      #pragma unroll
      for (int j = 0; j < 2; ++j)
        acc[i][j] = __builtin_amdgcn_wmma_f32_16x16x32_bf16(
            false, af[i].v, false, bfr[j].v, (short)0, acc[i][j], false, false);
    __syncthreads();
  }

  // Epilogue: C-layout element (m = tile_m + v + 8*(lane>=16), n = tile_n + lane&15)
  const int half = lane >> 4, nl = lane & 15;
  #pragma unroll
  for (int i = 0; i < 4; ++i) {
    #pragma unroll
    for (int j = 0; j < 2; ++j) {
      const int nt = n0 + wn * 32 + j * 16 + nl;
      const float bn = bias[nt];
      #pragma unroll
      for (int v = 0; v < 8; ++v) {
        const int mt = m0 + wm * 64 + i * 16 + v + 8 * half;
        const float val = acc[i][j][v] + bn;
        if (EPI == EPI_QKV) {
          const int sect = nt >> 10, w = nt & 1023, h = w >> 6, d = w & 63;
          const int b = mt >> 11, t = mt & 2047;
          const size_t bh = (size_t)(b * 16 + h);
          const uint16_t bv = f2bf(val);
          if (sect == 0)      oQ[(bh * 2048 + t) * 64 + d] = bv;
          else if (sect == 1) oK[(bh * 2048 + t) * 64 + d] = bv;
          else                oV[(bh * 64 + d) * 2048 + t] = bv;   // V transposed
        } else if (EPI == EPI_RES_F32) {
          outF[(size_t)mt * N + nt] = val + res[(size_t)mt * N + nt];
        } else { // exact-erf GELU, bf16 out
          const float gx = 0.5f * val * (1.0f + erff(val * 0.70710678118654752f));
          oQ[(size_t)mt * N + nt] = f2bf(gx);
        }
      }
    }
  }
}

// ---------------------------------------------------------------------------
// Causal flash attention: one block = (b, h, 128-query tile); 8 waves x 16 rows
// ---------------------------------------------------------------------------
__global__ __launch_bounds__(256)
void flash_kernel(const uint16_t* __restrict__ Q, const uint16_t* __restrict__ Kb,
                  const uint16_t* __restrict__ Vt, uint16_t* __restrict__ Y) {
  const int T = 2048;
  __shared__ uint32_t sK[64 * 36];       // 64 keys x 64 d (bf16), stride 36 dw
  __shared__ uint32_t sV[64 * 36];       // 64 d x 64 keys (V^T)
  __shared__ uint32_t sP[8 * 16 * 36];   // per-wave P staging (16 q x 64 keys)
  const int tid = threadIdx.x, lane = tid & 31, wave = tid >> 5;
  const int qb = blockIdx.x, bh = blockIdx.y;
  const int b = bh >> 4, h = bh & 15;
  const int Q0 = qb * 128;
  const int half = lane >> 4, mr = lane & 15, nl = lane & 15;
  uint32_t* pbuf = sP + wave * 16 * 36;

  // Q fragments (A operand, 16 rows x 64 d = 2 k-chunks), kept in registers
  const uint16_t* qrow = Q + ((size_t)bh * T + Q0 + wave * 16 + mr) * 64;
  Frag16 qf[2];
  #pragma unroll
  for (int c = 0; c < 2; ++c) {
    qf[c].u[0] = *(const uint4*)(qrow + c * 32 + half * 8);
    qf[c].u[1] = *(const uint4*)(qrow + c * 32 + 16 + half * 8);
  }

  v8f o[4] = {};
  float m_i[8], l_i[8];
  #pragma unroll
  for (int v = 0; v < 8; ++v) { m_i[v] = -1e30f; l_i[v] = 0.f; }

  const int kbEnd = 2 * qb + 1;                    // keys <= Q0+127
  for (int kb = 0; kb <= kbEnd; ++kb) {
    const int K0 = kb * 64;
    #pragma unroll
    for (int p = 0; p < 2; ++p) {
      const int it = tid + p * 256;
      const int r = it >> 3, c8 = it & 7;
      *(uint4*)&sK[r * 36 + c8 * 4] =
          *(const uint4*)(Kb + ((size_t)bh * T + K0 + r) * 64 + c8 * 8);
      *(uint4*)&sV[r * 36 + c8 * 4] =
          *(const uint4*)(Vt + ((size_t)bh * 64 + r) * T + K0 + c8 * 8);
    }
    __syncthreads();

    if (K0 <= Q0 + wave * 16 + 15) {   // wave-uniform guard: EXEC stays full
      // S = Q K^T  (4 key sub-tiles x 2 d-chunks)
      v8f s[4] = {};
      #pragma unroll
      for (int j = 0; j < 4; ++j)
        #pragma unroll
        for (int c = 0; c < 2; ++c) {
          Frag16 kf = ldsFragB(sK, j * 16, 36, lane, c * 16);
          s[j] = __builtin_amdgcn_wmma_f32_16x16x32_bf16(
              false, qf[c].v, false, kf.v, (short)0, s[j], false, false);
        }
      // online softmax; C-layout row = v + 8*half, col = j*16 + nl
      #pragma unroll
      for (int v = 0; v < 8; ++v) {
        const int qi = Q0 + wave * 16 + v + 8 * half;
        float mx = -1e30f;
        #pragma unroll
        for (int j = 0; j < 4; ++j) {
          const int ki = K0 + j * 16 + nl;
          const float sv = (ki <= qi) ? s[j][v] * 0.125f : -1e30f;
          s[j][v] = sv;
          mx = fmaxf(mx, sv);
        }
        #pragma unroll
        for (int off = 1; off < 16; off <<= 1) mx = fmaxf(mx, __shfl_xor(mx, off, 32));
        const float mo = m_i[v];
        const float mn = fmaxf(mo, mx);
        const float f = expf(mo - mn);
        float rs = 0.f;
        #pragma unroll
        for (int j = 0; j < 4; ++j) {
          const float pv = expf(s[j][v] - mn);
          s[j][v] = pv;
          rs += pv;
        }
        #pragma unroll
        for (int off = 1; off < 16; off <<= 1) rs += __shfl_xor(rs, off, 32);
        m_i[v] = mn;
        l_i[v] = l_i[v] * f + rs;
        #pragma unroll
        for (int jj = 0; jj < 4; ++jj) o[jj][v] *= f;
        // stage P row (bf16) into per-wave LDS buffer
        uint16_t* pr = (uint16_t*)(pbuf + (v + 8 * half) * 36);
        #pragma unroll
        for (int j = 0; j < 4; ++j) pr[j * 16 + nl] = f2bf(s[j][v]);
      }
      __asm__ volatile("" ::: "memory");   // same-wave LDS ops are in-order in HW
      // O += P V  (2 key-chunks x 4 d-tiles)
      #pragma unroll
      for (int c = 0; c < 2; ++c) {
        Frag16 pf = ldsFragA(pbuf, 0, 36, lane, c * 16);
        #pragma unroll
        for (int jj = 0; jj < 4; ++jj) {
          Frag16 vf = ldsFragB(sV, jj * 16, 36, lane, c * 16);
          o[jj] = __builtin_amdgcn_wmma_f32_16x16x32_bf16(
              false, pf.v, false, vf.v, (short)0, o[jj], false, false);
        }
      }
    }
    __syncthreads();
  }

  // normalize and write y[b*T+t][h*64+d] as bf16 (input to proj GEMM)
  #pragma unroll
  for (int v = 0; v < 8; ++v) {
    const float linv = 1.0f / l_i[v];
    const int t = Q0 + wave * 16 + v + 8 * half;
    const size_t rowo = ((size_t)b * T + t) * 1024 + h * 64;
    #pragma unroll
    for (int jj = 0; jj < 4; ++jj)
      Y[rowo + jj * 16 + nl] = f2bf(o[jj][v] * linv);
  }
}

// ---------------------------------------------------------------------------
// Launch
// ---------------------------------------------------------------------------
extern "C" void kernel_launch(void* const* d_in, const int* in_sizes, int n_in,
                              void* d_out, int out_size, void* d_ws, size_t ws_size,
                              hipStream_t stream) {
  const float* x     = (const float*)d_in[0];
  const float* ln1g  = (const float*)d_in[1];
  const float* ln1b  = (const float*)d_in[2];
  const float* wattn = (const float*)d_in[3];
  const float* battn = (const float*)d_in[4];
  const float* wproj = (const float*)d_in[5];
  const float* bproj = (const float*)d_in[6];
  const float* ln2g  = (const float*)d_in[7];
  const float* ln2b  = (const float*)d_in[8];
  const float* wfc   = (const float*)d_in[9];
  const float* bfc   = (const float*)d_in[10];
  const float* wfc2  = (const float*)d_in[11];
  const float* bfc2  = (const float*)d_in[12];

  const int Mr = 4096, C = 1024, C3 = 3072, C4 = 4096;

  char* wsp = (char*)d_ws;
  auto alloc = [&](size_t bytes) -> void* {
    void* p = (void*)wsp;
    wsp += (bytes + 255) & ~(size_t)255;
    return p;
  };
  uint16_t* wtA  = (uint16_t*)alloc((size_t)C3 * C * 2);   // w_attn^T bf16 [3C][C]
  uint16_t* wtP  = (uint16_t*)alloc((size_t)C  * C * 2);   // w_proj^T
  uint16_t* wtF  = (uint16_t*)alloc((size_t)C4 * C * 2);   // w_fc^T
  uint16_t* wtF2 = (uint16_t*)alloc((size_t)C  * C4 * 2);  // w_fc2^T
  uint16_t* ln1o = (uint16_t*)alloc((size_t)Mr * C * 2);
  uint16_t* qB   = (uint16_t*)alloc((size_t)Mr * C * 2);   // [b,h,t,d]
  uint16_t* kB   = (uint16_t*)alloc((size_t)Mr * C * 2);   // [b,h,t,d]
  uint16_t* vtB  = (uint16_t*)alloc((size_t)Mr * C * 2);   // [b,h,d,t]
  uint16_t* yB   = (uint16_t*)alloc((size_t)Mr * C * 2);
  float*    x1   = (float*)   alloc((size_t)Mr * C * 4);
  uint16_t* ln2o = (uint16_t*)alloc((size_t)Mr * C * 2);
  uint16_t* hB   = (uint16_t*)alloc((size_t)Mr * C4 * 2);

  const dim3 tb(32, 8);
  transpose_bf16_kernel<<<dim3(C3 / 32, C / 32),  tb, 0, stream>>>(wattn, wtA,  C,  C3);
  transpose_bf16_kernel<<<dim3(C  / 32, C / 32),  tb, 0, stream>>>(wproj, wtP,  C,  C);
  transpose_bf16_kernel<<<dim3(C4 / 32, C / 32),  tb, 0, stream>>>(wfc,   wtF,  C,  C4);
  transpose_bf16_kernel<<<dim3(C  / 32, C4 / 32), tb, 0, stream>>>(wfc2,  wtF2, C4, C);

  ln_bf16_kernel<<<Mr, 256, 0, stream>>>(x, ln1g, ln1b, ln1o);

  gemm_bf16_kernel<EPI_QKV><<<dim3(C3 / 128, Mr / 128), 256, 0, stream>>>(
      ln1o, wtA, battn, Mr, C3, C, nullptr, nullptr, qB, kB, vtB);

  flash_kernel<<<dim3(16, 32), 256, 0, stream>>>(qB, kB, vtB, yB);

  gemm_bf16_kernel<EPI_RES_F32><<<dim3(C / 128, Mr / 128), 256, 0, stream>>>(
      yB, wtP, bproj, Mr, C, C, x1, x, nullptr, nullptr, nullptr);

  ln_bf16_kernel<<<Mr, 256, 0, stream>>>(x1, ln2g, ln2b, ln2o);

  gemm_bf16_kernel<EPI_GELU_BF16><<<dim3(C4 / 128, Mr / 128), 256, 0, stream>>>(
      ln2o, wtF, bfc, Mr, C4, C, nullptr, nullptr, hB, nullptr, nullptr);

  gemm_bf16_kernel<EPI_RES_F32><<<dim3(C / 128, Mr / 128), 256, 0, stream>>>(
      hB, wtF2, bfc2, Mr, C, C4, (float*)d_out, x1, nullptr, nullptr, nullptr);

  (void)in_sizes; (void)n_in; (void)out_size; (void)ws_size;
}
